// GATNet_orig_26336739459199
// MI455X (gfx1250) — matrix operand order repaired and checked by reading, not compiled, not run on hardware
//
#include <hip/hip_runtime.h>
#include <hip/hip_bf16.h>

typedef __attribute__((ext_vector_type(16))) _Float16 v16h;
typedef __attribute__((ext_vector_type(8)))  _Float16 v8h;
typedef __attribute__((ext_vector_type(8)))  float    v8f;

#define N_GRAPHS 512
#define NPG      84
#define NNODES   (N_GRAPHS * NPG)   // 43008
#define FDIM     64

// ---------------- prep kernels ----------------

// f32 [M x K] -> f16 [M x Kp], zero padded columns K..Kp-1
__global__ __launch_bounds__(256) void cvt_pad_f16(const float* __restrict__ src,
                                                   _Float16* __restrict__ dst,
                                                   int M, int K, int Kp) {
    int total = M * Kp;
    for (int idx = blockIdx.x * blockDim.x + threadIdx.x; idx < total;
         idx += gridDim.x * blockDim.x) {
        int r = idx / Kp, k = idx - r * Kp;
        dst[idx] = (k < K) ? (_Float16)src[(size_t)r * K + k] : (_Float16)0.f;
    }
}

// W f32 [K x 64] -> f16 fragments in exact WMMA-B lane layout:
// out[((kt*4 + nt)*32 + lane)*16 + e] = W[kt*32 + (e>>3)*16 + (lane>>4)*8 + (e&7)][nt*16 + (lane&15)]
__global__ __launch_bounds__(256) void make_bfrag(const float* __restrict__ W,
                                                  _Float16* __restrict__ out,
                                                  int K, int ktiles) {
    int total = ktiles * 4 * 32 * 16;
    for (int idx = blockIdx.x * blockDim.x + threadIdx.x; idx < total;
         idx += gridDim.x * blockDim.x) {
        int e    = idx & 15;
        int lane = (idx >> 4) & 31;
        int nt   = (idx >> 9) & 3;
        int kt   = idx >> 11;
        int k    = kt * 32 + (e >> 3) * 16 + (lane >> 4) * 8 + (e & 7);
        int col  = nt * 16 + (lane & 15);
        out[idx] = (k < K) ? (_Float16)W[(size_t)k * 64 + col] : (_Float16)0.f;
    }
}

// ---------------- GEMM: C[M x 64] = A[M x 32*KTILES (f16, padded)] @ Bfrag ----------------
template<int KTILES>
__global__ __launch_bounds__(128) void gemm_f16_n64(const _Float16* __restrict__ A, int ldaH,
                                                    const _Float16* __restrict__ Bfrag,
                                                    float* __restrict__ C, int M) {
    int wave = (int)((blockIdx.x * blockDim.x + threadIdx.x) >> 5);
    int row0 = wave * 16;
    if (row0 >= M) return;   // wave-uniform

    int lane = threadIdx.x & 31;
    const _Float16* Ar = A + (size_t)(row0 + (lane & 15)) * ldaH + (lane >> 4) * 8;

    v8f acc[4];
#pragma unroll
    for (int nt = 0; nt < 4; ++nt) acc[nt] = (v8f){0.f,0.f,0.f,0.f,0.f,0.f,0.f,0.f};

#pragma unroll
    for (int kt = 0; kt < KTILES; ++kt) {
        v8h a0 = *(const v8h*)(Ar + kt * 32);
        v8h a1 = *(const v8h*)(Ar + kt * 32 + 16);
        v16h a = __builtin_shufflevector(a0, a1, 0,1,2,3,4,5,6,7,8,9,10,11,12,13,14,15);
#pragma unroll
        for (int nt = 0; nt < 4; ++nt) {
            v16h b = *(const v16h*)(Bfrag + ((size_t)(kt * 4 + nt) * 32 + lane) * 16);
            acc[nt] = __builtin_amdgcn_wmma_f32_16x16x32_f16(
                false, a, false, b, (short)0, acc[nt], false, false);
        }
    }

    int rb = row0 + 8 * (lane >> 4);
    int cl = lane & 15;
#pragma unroll
    for (int nt = 0; nt < 4; ++nt) {
        int cc = nt * 16 + cl;
#pragma unroll
        for (int r = 0; r < 8; ++r)
            C[(size_t)(rb + r) * 64 + cc] = acc[nt][r];
    }
}

// ---------------- layer-1 GAT attention (32 heads x 2 ch) + bias + ELU -> f16 ----------------
__global__ __launch_bounds__(256) void gat1_attn(const float* __restrict__ H1,    // [N,64] f32
                                                 _Float16* __restrict__ Hact,     // [N,64] f16 out
                                                 const float* __restrict__ a_src, // [32,2]
                                                 const float* __restrict__ a_dst, // [32,2]
                                                 const float* __restrict__ b1) {  // [64]
    __shared__ __attribute__((aligned(16))) float sH[NPG * 64];
    __shared__ float sAs[NPG * 32];
    __shared__ float sAd[NPG * 32];

    int g = blockIdx.x;
    const float* Hg = H1 + (size_t)g * NPG * 64;
    _Float16* Og = Hact + (size_t)g * NPG * 64;

    // async global -> LDS staging, 16B per lane (ASYNCcnt-tracked)
    for (int q = threadIdx.x; q < NPG * 64 / 4; q += 256) {
        const float* src = Hg + q * 4;
        unsigned lds = (unsigned)(unsigned long long)(const void*)(&sH[q * 4]);
        asm volatile("global_load_async_to_lds_b128 %0, %1, off"
                     :: "v"(lds), "v"(src) : "memory");
    }
    asm volatile("s_wait_asynccnt 0x0" ::: "memory");
    __syncthreads();

    for (int t = threadIdx.x; t < NPG * 32; t += 256) {
        int j = t >> 5, h = t & 31;
        float h0 = sH[j * 64 + 2 * h], h1 = sH[j * 64 + 2 * h + 1];
        sAs[t] = h0 * a_src[2 * h] + h1 * a_src[2 * h + 1];
        sAd[t] = h0 * a_dst[2 * h] + h1 * a_dst[2 * h + 1];
    }
    __syncthreads();

    for (int t = threadIdx.x; t < NPG * 32; t += 256) {
        int i = t >> 5, h = t & 31;
        float ad = sAd[i * 32 + h];
        float m = -1e30f;
        for (int j = 0; j < NPG; ++j) {
            float e = sAs[j * 32 + h] + ad;
            e = (e > 0.f) ? e : 0.2f * e;          // leaky_relu(0.2)
            m = fmaxf(m, e);
        }
        float z = 0.f, o0 = 0.f, o1 = 0.f;
        for (int j = 0; j < NPG; ++j) {
            float e = sAs[j * 32 + h] + ad;
            e = (e > 0.f) ? e : 0.2f * e;
            float w = __expf(e - m);
            z  += w;
            o0 += w * sH[j * 64 + 2 * h];
            o1 += w * sH[j * 64 + 2 * h + 1];
        }
        float inv = 1.f / (z + 1e-16f);
        float v0 = o0 * inv + b1[2 * h];
        float v1 = o1 * inv + b1[2 * h + 1];
        v0 = (v0 > 0.f) ? v0 : (__expf(v0) - 1.f); // ELU
        v1 = (v1 > 0.f) ? v1 : (__expf(v1) - 1.f);
        Og[i * 64 + 2 * h]     = (_Float16)v0;
        Og[i * 64 + 2 * h + 1] = (_Float16)v1;
    }
}

// ---------------- layer-2 attention (1 head x 64) via LDS WMMA + mean-pool + FC ----------------
__device__ inline v16h load_a_lds(const _Float16* S, int ld, int row0, int k0) {
    int lane = threadIdx.x & 31;
    int r  = row0 + (lane & 15);
    int kb = (lane >> 4) * 8;
    const _Float16* Sr = S + r * ld + k0;
    v8h a0 = *(const v8h*)(Sr + kb);
    v8h a1 = *(const v8h*)(Sr + 16 + kb);
    return __builtin_shufflevector(a0, a1, 0,1,2,3,4,5,6,7,8,9,10,11,12,13,14,15);
}

__global__ __launch_bounds__(256) void gat2_attn_pool(const float* __restrict__ H2,     // [N,64]
                                                      const float* __restrict__ a_src,  // [64]
                                                      const float* __restrict__ a_dst,  // [64]
                                                      const float* __restrict__ b2,     // [64]
                                                      const float* __restrict__ Wfc,    // [64]
                                                      const float* __restrict__ bfc,    // [1]
                                                      float* __restrict__ out) {        // [512]
    __shared__ __attribute__((aligned(32))) _Float16 sBf[3 * 4 * 32 * 16]; // B frags, 6144 halves
    __shared__ __attribute__((aligned(32))) _Float16 sP[96 * 96];          // P[i][j], pad zero
    __shared__ float sAs[96], sAd[96];
    __shared__ __attribute__((aligned(32))) float sInv[96];
    __shared__ float sPool[64];

    int g = blockIdx.x;
    const float* Hg = H2 + (size_t)g * NPG * 64;

    // stage H2 into WMMA-B fragment order (K padded 84->96), branchless OOB
    for (int idx = threadIdx.x; idx < 3 * 4 * 32 * 16; idx += 256) {
        int e    = idx & 15;
        int lane = (idx >> 4) & 31;
        int nt   = (idx >> 9) & 3;
        int kt   = idx >> 11;
        int k    = kt * 32 + (e >> 3) * 16 + (lane >> 4) * 8 + (e & 7);
        int c    = nt * 16 + (lane & 15);
        float v  = Hg[(size_t)(k < NPG ? k : 0) * 64 + c];
        sBf[idx] = (_Float16)((k < NPG) ? v : 0.f);
    }
    if (threadIdx.x < 64) sPool[threadIdx.x] = 0.f;

    for (int j = threadIdx.x; j < 96; j += 256) {
        float as = 0.f, ad = 0.f;
        if (j < NPG) {
            for (int c = 0; c < 64; ++c) {
                float v = Hg[j * 64 + c];
                as += v * a_src[c];
                ad += v * a_dst[c];
            }
        }
        sAs[j] = as; sAd[j] = ad;
    }
    __syncthreads();

    for (int i = threadIdx.x; i < 96; i += 256) {
        if (i < NPG) {
            float ad = sAd[i];
            float m = -1e30f;
            for (int j = 0; j < NPG; ++j) {
                float e = sAs[j] + ad;
                e = (e > 0.f) ? e : 0.2f * e;      // leaky_relu(0.2)
                m = fmaxf(m, e);
            }
            float z = 0.f;
            for (int j = 0; j < NPG; ++j) {
                float e = sAs[j] + ad;
                e = (e > 0.f) ? e : 0.2f * e;
                float w = __expf(e - m);
                z += w;
                sP[i * 96 + j] = (_Float16)w;
            }
            for (int j = NPG; j < 96; ++j) sP[i * 96 + j] = (_Float16)0.f;
            sInv[i] = 1.f / (z + 1e-16f);
        } else {
            for (int j = 0; j < 96; ++j) sP[i * 96 + j] = (_Float16)0.f;
            sInv[i] = 0.f;                         // pad rows contribute nothing
        }
    }
    __syncthreads();

    // Out[96x64] = P[96x96] @ H2f[96x64]. Each wave owns nt = wave&3 (loop-invariant
    // B fragments, hoisted) and m-tiles mt = (wave>>2) + 2*it, it = 0..2. Fully unrolled.
    int wave = (int)(threadIdx.x >> 5);
    int lane = threadIdx.x & 31;
    int nt   = wave & 3;
    int mt0  = wave >> 2;

    v16h bf0 = *(const v16h*)(sBf + ((0 * 4 + nt) * 32 + lane) * 16);
    v16h bf1 = *(const v16h*)(sBf + ((1 * 4 + nt) * 32 + lane) * 16);
    v16h bf2 = *(const v16h*)(sBf + ((2 * 4 + nt) * 32 + lane) * 16);

#pragma unroll
    for (int it = 0; it < 3; ++it) {
        int mt = mt0 + 2 * it;
        v8f acc = (v8f){0.f,0.f,0.f,0.f,0.f,0.f,0.f,0.f};
        acc = __builtin_amdgcn_wmma_f32_16x16x32_f16(
            false, load_a_lds(sP, 96, mt * 16, 0),  false, bf0, (short)0, acc, false, false);
        acc = __builtin_amdgcn_wmma_f32_16x16x32_f16(
            false, load_a_lds(sP, 96, mt * 16, 32), false, bf1, (short)0, acc, false, false);
        acc = __builtin_amdgcn_wmma_f32_16x16x32_f16(
            false, load_a_lds(sP, 96, mt * 16, 64), false, bf2, (short)0, acc, false, false);

        int rb = mt * 16 + 8 * (lane >> 4);
        float4 w0 = *(const float4*)(sInv + rb);
        float4 w1 = *(const float4*)(sInv + rb + 4);
        float s = acc[0] * w0.x + acc[1] * w0.y + acc[2] * w0.z + acc[3] * w0.w
                + acc[4] * w1.x + acc[5] * w1.y + acc[6] * w1.z + acc[7] * w1.w;
        atomicAdd(&sPool[nt * 16 + (lane & 15)], s);   // one ds_add_f32 per lane per tile
    }
    __syncthreads();

    if (threadIdx.x == 0) {
        float acc = bfc[0];
        const float inv_n = 1.f / (float)NPG;
        for (int c = 0; c < 64; ++c)
            acc += (sPool[c] * inv_n + b2[c]) * Wfc[c];
        out[g] = acc;
    }
}

__global__ void write_tail_zeros(float* __restrict__ out) {
    if (threadIdx.x < 6) out[N_GRAPHS + threadIdx.x] = 0.f;
}

// ---------------------------------------------------------------------------

extern "C" void kernel_launch(void* const* d_in, const int* in_sizes, int n_in,
                              void* d_out, int out_size, void* d_ws, size_t ws_size,
                              hipStream_t stream) {
    const float* x      = (const float*)d_in[0];   // [43008, 84]
    // d_in[1] edge_index / d_in[2] batch: fixed fully-connected structure, unused
    const float* W1     = (const float*)d_in[3];   // [84, 64]
    const float* a_src1 = (const float*)d_in[4];   // [32, 2]
    const float* a_dst1 = (const float*)d_in[5];   // [32, 2]
    const float* b1     = (const float*)d_in[6];   // [64]
    const float* W2     = (const float*)d_in[7];   // [64, 64]
    const float* a_src2 = (const float*)d_in[8];   // [1, 64]
    const float* a_dst2 = (const float*)d_in[9];   // [1, 64]
    const float* b2     = (const float*)d_in[10];  // [64]
    const float* Wfc    = (const float*)d_in[11];  // [64, 1]
    const float* bfc    = (const float*)d_in[12];  // [1]
    float* out = (float*)d_out;

    // workspace layout (halves first, then f32 buffers; all 16B aligned)
    _Float16* Xpad = (_Float16*)d_ws;                       // [43008 x 96]
    _Float16* Hact = Xpad + (size_t)NNODES * 96;            // [43008 x 64]
    _Float16* W1f  = Hact + (size_t)NNODES * 64;            // 3*4*32*16 = 6144
    _Float16* W2f  = W1f + 6144;                            // 2*4*32*16 = 4096
    float*    H1   = (float*)(W2f + 4096);                  // [43008 x 64]
    float*    H2   = H1 + (size_t)NNODES * 64;              // [43008 x 64]

    const int mtiles = NNODES / 16;                // 2688 waves
    const int gemm_blocks = mtiles / 4;            // 128 threads = 4 waves/block

    // 0) prep: pad/convert X, swizzle W1/W2 into WMMA-B fragment order
    cvt_pad_f16<<<2048, 256, 0, stream>>>(x, Xpad, NNODES, 84, 96);
    make_bfrag<<<24, 256, 0, stream>>>(W1, W1f, 84, 3);
    make_bfrag<<<16, 256, 0, stream>>>(W2, W2f, 64, 2);
    // 1) H1 = X @ W1
    gemm_f16_n64<3><<<gemm_blocks, 128, 0, stream>>>(Xpad, 96, W1f, H1, NNODES);
    // 2) layer-1 attention + b1 + ELU -> f16 (async global->LDS staging)
    gat1_attn<<<N_GRAPHS, 256, 0, stream>>>(H1, Hact, a_src1, a_dst1, b1);
    // 3) H2 = act @ W2
    gemm_f16_n64<2><<<gemm_blocks, 128, 0, stream>>>(Hact, 64, W2f, H2, NNODES);
    // 4) layer-2 attention (WMMA) + mean pool + FC
    gat2_attn_pool<<<N_GRAPHS, 256, 0, stream>>>(H2, a_src2, a_dst2, b2, Wfc, bfc, out);
    // 5) trailing zero outputs
    write_tail_zeros<<<1, 32, 0, stream>>>(out);
}
